// MultiHeadAttention_71159018160989
// MI455X (gfx1250) — compile-verified, matrix operand order checked by reading, not run on hardware
//
#include <hip/hip_runtime.h>

// ---------------------------------------------------------------------------
// MI455X (gfx1250) multi-head attention forward, wave32 + WMMA bf16.
//   B=4, S=1024, D=1024, H=16, DH=64;  reshape-view: [64 batches][1024 seq][64 hd]
//   d_out = [4M floats output][64M floats attn]
//   d_ws (bf16 elements):
//     Xqb(4M) Xkb(4M) Xvb(4M) Wqb(1M) Wkb(1M) Wvb(1M) Qb(4M) Kb(4M) Vtb(4M)
//     = 27M bf16 = 54MB
// ---------------------------------------------------------------------------

typedef __attribute__((ext_vector_type(16))) __bf16 v16bf;
typedef __attribute__((ext_vector_type(8)))  float  v8f;

#define WMMA_BF16(a, b, c) \
  __builtin_amdgcn_wmma_f32_16x16x32_bf16(false, (a), false, (b), (short)0, (c), false, false)

union FragB16 { v16bf v; uint4 q[2]; unsigned short h[16]; };
union Acc8    { v8f v; float f[8]; };

static __device__ __forceinline__ unsigned short f32_to_bf16(float f) {
  unsigned int u = __float_as_uint(f);
  u += 0x7fffu + ((u >> 16) & 1u);          // round-to-nearest-even
  return (unsigned short)(u >> 16);
}

static __device__ __forceinline__ unsigned int pack_bf16x2(float lo, float hi) {
  return ((unsigned int)f32_to_bf16(hi) << 16) | (unsigned int)f32_to_bf16(lo);
}

// A/B fragment (16x32 bf16, wave32 layout): lane L holds row (L&15),
// k-chunks [ko, ko+8) and [ko+16, ko+24) with ko = kbase + 8*(L>>4).
static __device__ __forceinline__ v16bf load_frag_bf16(const unsigned short* __restrict__ base,
                                                       int ld, int kbase, int lane) {
  const int r  = lane & 15;
  const int ko = kbase + ((lane >> 4) << 3);
  FragB16 f;
  f.q[0] = *(const uint4*)(base + (size_t)r * ld + ko);
  f.q[1] = *(const uint4*)(base + (size_t)r * ld + ko + 16);
  return f.v;
}

// ---------------------------------------------------------------------------
// Kernel 0: bulk f32 -> bf16 conversion with packed b128 stores.
// One thread handles 8 elements (2x float4 in, 1x uint4 out).
// ---------------------------------------------------------------------------
__global__ __launch_bounds__(256)
void cvt_bf16_kernel(const float* __restrict__ src, unsigned int* __restrict__ dst) {
  const size_t t = (size_t)blockIdx.x * 256 + threadIdx.x;
  const float4* p = (const float4*)src + t * 2;
  const float4 a = p[0], b = p[1];
  uint4 o;
  o.x = pack_bf16x2(a.x, a.y);
  o.y = pack_bf16x2(a.z, a.w);
  o.z = pack_bf16x2(b.x, b.y);
  o.w = pack_bf16x2(b.z, b.w);
  ((uint4*)dst)[t] = o;
}

// ---------------------------------------------------------------------------
// Kernel 1: QKV projection.  out[m,n] = sum_k in[m,k]*W[n,k] + b[n]
// All operands pre-converted bf16, so the k-loop is pure b128 loads + WMMA.
// grid (128 Mtiles-of-32, 2 Nhalves, 3 projections), block 256 (8 waves).
// Wave computes 32x64 (2 Mtiles x 4 Ntiles): B frags reused 2x, A frags 4x.
// Q,K stored row-major bf16; V stored per-batch transposed (Vt[bh][dh][s]).
// ---------------------------------------------------------------------------
__global__ __launch_bounds__(256)
void mha_qkv_gemm_kernel(const unsigned short* __restrict__ Xq,
                         const unsigned short* __restrict__ Xk,
                         const unsigned short* __restrict__ Xv,
                         const unsigned short* __restrict__ Wqb,
                         const unsigned short* __restrict__ Wkb,
                         const unsigned short* __restrict__ Wvb,
                         const float* __restrict__ bq, const float* __restrict__ bk,
                         const float* __restrict__ bv,
                         unsigned short* __restrict__ Qb, unsigned short* __restrict__ Kb,
                         unsigned short* __restrict__ Vtb) {
  const int lane  = threadIdx.x & 31;
  const int w     = threadIdx.x >> 5;
  const int mbase = blockIdx.x * 32;
  const int ncb   = blockIdx.y * 512 + w * 64;
  const int z     = blockIdx.z;

  const unsigned short* X   = (z == 0) ? Xq  : (z == 1) ? Xk  : Xv;
  const unsigned short* W   = (z == 0) ? Wqb : (z == 1) ? Wkb : Wvb;
  const float*          bia = (z == 0) ? bq  : (z == 1) ? bk  : bv;

  Acc8 acc[2][4];
  #pragma unroll
  for (int mt = 0; mt < 2; ++mt)
    #pragma unroll
    for (int t = 0; t < 4; ++t)
      #pragma unroll
      for (int r = 0; r < 8; ++r) acc[mt][t].f[r] = 0.0f;

  const unsigned short* arow0 = X + (size_t)mbase * 1024;
  const unsigned short* arow1 = X + (size_t)(mbase + 16) * 1024;
  for (int k = 0; k < 1024; k += 32) {
    v16bf a0 = load_frag_bf16(arow0, 1024, k, lane);
    v16bf a1 = load_frag_bf16(arow1, 1024, k, lane);
    __builtin_prefetch(arow0 + (size_t)(lane & 15) * 1024 + k + 32, 0, 3);
    #pragma unroll
    for (int t = 0; t < 4; ++t) {
      v16bf b = load_frag_bf16(W + (size_t)(ncb + t * 16) * 1024, 1024, k, lane);
      acc[0][t].v = WMMA_BF16(a0, b, acc[0][t].v);
      acc[1][t].v = WMMA_BF16(a1, b, acc[1][t].v);
    }
  }

  const int mloc = (lane >> 4) << 3;   // C layout: lanes<16 rows 0..7, lanes>=16 rows 8..15
  #pragma unroll
  for (int t = 0; t < 4; ++t) {
    const int n  = ncb + t * 16 + (lane & 15);
    const float bb = bia[n];
    #pragma unroll
    for (int mt = 0; mt < 2; ++mt) {
      #pragma unroll
      for (int r = 0; r < 8; ++r) {
        const int m = mbase + mt * 16 + mloc + r;
        const unsigned short hv = f32_to_bf16(acc[mt][t].f[r] + bb);
        if (z == 0)      Qb[(size_t)m * 1024 + n] = hv;
        else if (z == 1) Kb[(size_t)m * 1024 + n] = hv;
        else {
          const unsigned int e  = (unsigned int)m * 1024u + (unsigned int)n;
          const unsigned int bh = e >> 16, s2 = (e >> 6) & 1023u, dh = e & 63u;
          Vtb[(size_t)bh * 65536 + (size_t)dh * 1024 + s2] = hv;
        }
      }
    }
  }
}

// ---------------------------------------------------------------------------
// Kernel 2: fused attention for one (batch, 16-query-row block).
// grid (64 qblocks, 64 batches), block 256 (8 waves).
// Wave w handles keys [128w, 128w+128): scores via 2 WMMA per 16-key tile,
// cross-wave softmax via shfl_xor + LDS, attn written once to HBM,
// bf16 probabilities staged in LDS, context = P@V via WMMA (K split 2-way).
// ---------------------------------------------------------------------------
__global__ __launch_bounds__(256)
void mha_attention_kernel(const unsigned short* __restrict__ Qb,
                          const unsigned short* __restrict__ Kb,
                          const unsigned short* __restrict__ Vtb,
                          float* __restrict__ out) {
  __shared__ unsigned short probs[16 * 1024];   // 32KB bf16 probabilities
  __shared__ float redbuf[8][16];               // softmax cross-wave stats
  __shared__ float cred[4][32][8];              // context partial reduction

  const int lane  = threadIdx.x & 31;
  const int w     = threadIdx.x >> 5;
  const int half  = lane >> 4;
  const int qbase = blockIdx.x * 16;
  const int bh    = blockIdx.y;

  const unsigned short* Qrow = Qb + (size_t)bh * 65536 + (size_t)qbase * 64;
  const unsigned short* Krow = Kb + (size_t)bh * 65536;

  v16bf aq0 = load_frag_bf16(Qrow, 64, 0, lane);
  v16bf aq1 = load_frag_bf16(Qrow, 64, 32, lane);

  // ---- scores: Q (16x64) x K^T (64x128 per wave) ----
  Acc8 acc[8];
  #pragma unroll
  for (int t = 0; t < 8; ++t) {
    const unsigned short* kr = Krow + (size_t)(w * 128 + t * 16) * 64;
    v16bf b0 = load_frag_bf16(kr, 64, 0, lane);
    v16bf b1 = load_frag_bf16(kr, 64, 32, lane);
    #pragma unroll
    for (int r = 0; r < 8; ++r) acc[t].f[r] = 0.0f;
    acc[t].v = WMMA_BF16(aq0, b0, acc[t].v);
    acc[t].v = WMMA_BF16(aq1, b1, acc[t].v);
  }

  // ---- row max (tiles -> 16-lane half -> cross-wave) ----
  float stat[8];
  #pragma unroll
  for (int r = 0; r < 8; ++r) {
    float m = acc[0].f[r];
    #pragma unroll
    for (int t = 1; t < 8; ++t) m = fmaxf(m, acc[t].f[r]);
    #pragma unroll
    for (int d = 8; d >= 1; d >>= 1) m = fmaxf(m, __shfl_xor(m, d, 32));
    stat[r] = m;
  }
  if ((lane & 15) == 0) {
    #pragma unroll
    for (int r = 0; r < 8; ++r) redbuf[w][half * 8 + r] = stat[r];
  }
  __syncthreads();
  float rowmax[8];
  #pragma unroll
  for (int r = 0; r < 8; ++r) {
    const int row = half * 8 + r;
    float m = redbuf[0][row];
    #pragma unroll
    for (int ww = 1; ww < 8; ++ww) m = fmaxf(m, redbuf[ww][row]);
    rowmax[r] = m;
  }
  __syncthreads();

  // ---- exp(scale*(x-max)) + row sum ----
  const float scale = 0.03125f;   // D**-0.5 = 1/32 (model dim, per reference)
  #pragma unroll
  for (int r = 0; r < 8; ++r) stat[r] = 0.0f;
  #pragma unroll
  for (int t = 0; t < 8; ++t) {
    #pragma unroll
    for (int r = 0; r < 8; ++r) {
      const float p = __expf((acc[t].f[r] - rowmax[r]) * scale);
      acc[t].f[r] = p;
      stat[r] += p;
    }
  }
  #pragma unroll
  for (int r = 0; r < 8; ++r) {
    #pragma unroll
    for (int d = 8; d >= 1; d >>= 1) stat[r] += __shfl_xor(stat[r], d, 32);
  }
  if ((lane & 15) == 0) {
    #pragma unroll
    for (int r = 0; r < 8; ++r) redbuf[w][half * 8 + r] = stat[r];
  }
  __syncthreads();
  float inv[8];
  #pragma unroll
  for (int r = 0; r < 8; ++r) {
    const int row = half * 8 + r;
    float s = redbuf[0][row];
    #pragma unroll
    for (int ww = 1; ww < 8; ++ww) s += redbuf[ww][row];
    inv[r] = 1.0f / s;
  }

  // ---- normalize; emit attn (f32, HBM, once) + probs (bf16, LDS) ----
  float* attn_out = out + (size_t)4 * 1024 * 1024
                  + (size_t)bh * 1024 * 1024 + (size_t)qbase * 1024;
  #pragma unroll
  for (int t = 0; t < 8; ++t) {
    const int col = w * 128 + t * 16 + (lane & 15);
    #pragma unroll
    for (int r = 0; r < 8; ++r) {
      const int row = half * 8 + r;
      const float p = acc[t].f[r] * inv[r];
      attn_out[(size_t)row * 1024 + col] = p;
      probs[row * 1024 + col] = f32_to_bf16(p);
    }
  }
  __syncthreads();

  // ---- context = P (16x1024) @ V (1024x64); wave -> (ntile, K-half) ----
  const int ntile = w & 3;
  const int khalf = w >> 2;
  const unsigned short* vrow = Vtb + (size_t)bh * 65536 + (size_t)(ntile * 16) * 1024;
  Acc8 ctx;
  #pragma unroll
  for (int r = 0; r < 8; ++r) ctx.f[r] = 0.0f;
  const int k0 = khalf * 512;
  for (int k = k0; k < k0 + 512; k += 32) {
    v16bf ap  = load_frag_bf16(&probs[0], 1024, k, lane);
    v16bf bvf = load_frag_bf16(vrow, 1024, k, lane);
    ctx.v = WMMA_BF16(ap, bvf, ctx.v);
  }
  if (khalf == 1) {
    #pragma unroll
    for (int r = 0; r < 8; ++r) cred[ntile][lane][r] = ctx.f[r];
  }
  __syncthreads();
  if (khalf == 0) {
    const int n = ntile * 16 + (lane & 15);
    #pragma unroll
    for (int r = 0; r < 8; ++r) {
      const float v = ctx.f[r] + cred[ntile][lane][r];
      const int row = half * 8 + r;
      out[(size_t)bh * 65536 + (size_t)(qbase + row) * 64 + n] = v;
    }
  }
}

// ---------------------------------------------------------------------------
// Kernel 3: out = LayerNorm(residual + context) * gamma + beta, in place.
// grid 4096 rows, block 256; 4 elements/thread; shfl + LDS reduction.
// ---------------------------------------------------------------------------
__global__ __launch_bounds__(256)
void mha_ln_residual_kernel(const float* __restrict__ residual,
                            const float* __restrict__ gamma,
                            const float* __restrict__ beta,
                            float* __restrict__ out) {
  __shared__ float ssum[8], ssq[8];
  const int tid  = threadIdx.x;
  const int lane = tid & 31, w = tid >> 5;
  const size_t base = (size_t)blockIdx.x * 1024;

  float x[4];
  float s = 0.0f, q = 0.0f;
  #pragma unroll
  for (int i = 0; i < 4; ++i) {
    const int c = tid + i * 256;
    x[i] = residual[base + c] + out[base + c];
    s += x[i];
    q += x[i] * x[i];
  }
  #pragma unroll
  for (int d = 16; d >= 1; d >>= 1) {
    s += __shfl_xor(s, d, 32);
    q += __shfl_xor(q, d, 32);
  }
  if (lane == 0) { ssum[w] = s; ssq[w] = q; }
  __syncthreads();
  float ts = 0.0f, tq = 0.0f;
  #pragma unroll
  for (int i = 0; i < 8; ++i) { ts += ssum[i]; tq += ssq[i]; }
  const float mean = ts * (1.0f / 1024.0f);
  const float var  = tq * (1.0f / 1024.0f) - mean * mean;
  const float rstd = rsqrtf(var + 1e-5f);
  #pragma unroll
  for (int i = 0; i < 4; ++i) {
    const int c = tid + i * 256;
    out[base + c] = (x[i] - mean) * rstd * gamma[c] + beta[c];
  }
}

// ---------------------------------------------------------------------------
extern "C" void kernel_launch(void* const* d_in, const int* in_sizes, int n_in,
                              void* d_out, int out_size, void* d_ws, size_t ws_size,
                              hipStream_t stream) {
  (void)in_sizes; (void)n_in; (void)out_size; (void)ws_size;
  const float* key   = (const float*)d_in[0];
  const float* value = (const float*)d_in[1];
  const float* query = (const float*)d_in[2];
  const float* Wq    = (const float*)d_in[3];
  const float* bq    = (const float*)d_in[4];
  const float* Wk    = (const float*)d_in[5];
  const float* bk    = (const float*)d_in[6];
  const float* Wv    = (const float*)d_in[7];
  const float* bv    = (const float*)d_in[8];
  const float* gamma = (const float*)d_in[9];
  const float* beta  = (const float*)d_in[10];
  float* out = (float*)d_out;

  const size_t MSZ = (size_t)4096 * 1024;  // activation elements (4M)
  const size_t WSZ = (size_t)1024 * 1024;  // weight elements (1M)
  unsigned short* Xqb = (unsigned short*)d_ws;   // bf16 activations
  unsigned short* Xkb = Xqb + MSZ;
  unsigned short* Xvb = Xkb + MSZ;
  unsigned short* Wqb = Xvb + MSZ;               // bf16 weights
  unsigned short* Wkb = Wqb + WSZ;
  unsigned short* Wvb = Wkb + WSZ;
  unsigned short* Qb  = Wvb + WSZ;               // projected bf16
  unsigned short* Kb  = Qb + MSZ;
  unsigned short* Vtb = Kb + MSZ;                // per-batch transposed V

  // f32 -> bf16 bulk conversions (8 elements/thread)
  cvt_bf16_kernel<<<dim3(2048), 256, 0, stream>>>(query, (unsigned int*)Xqb);
  cvt_bf16_kernel<<<dim3(2048), 256, 0, stream>>>(key,   (unsigned int*)Xkb);
  cvt_bf16_kernel<<<dim3(2048), 256, 0, stream>>>(value, (unsigned int*)Xvb);
  cvt_bf16_kernel<<<dim3(512),  256, 0, stream>>>(Wq,    (unsigned int*)Wqb);
  cvt_bf16_kernel<<<dim3(512),  256, 0, stream>>>(Wk,    (unsigned int*)Wkb);
  cvt_bf16_kernel<<<dim3(512),  256, 0, stream>>>(Wv,    (unsigned int*)Wvb);

  mha_qkv_gemm_kernel<<<dim3(128, 2, 3), 256, 0, stream>>>(
      Xqb, Xkb, Xvb, Wqb, Wkb, Wvb, bq, bk, bv, Qb, Kb, Vtb);
  mha_attention_kernel<<<dim3(64, 64), 256, 0, stream>>>(Qb, Kb, Vtb, out);
  mha_ln_residual_kernel<<<dim3(4096), 256, 0, stream>>>(query, gamma, beta, out);
}
